// ResidualCodebook_33535104647889
// MI455X (gfx1250) — matrix operand order repaired and checked by reading, not compiled, not run on hardware
//
#include <hip/hip_runtime.h>

typedef __attribute__((ext_vector_type(16))) _Float16 v16h;
typedef __attribute__((ext_vector_type(8)))  float    v8f;
typedef __attribute__((ext_vector_type(2)))  float    v2f;

#define NTOK    32768      // B*S = 16*2048
#define INDIM   512
#define DLAT    64
#define KCODE   4096
#define HSTEPS  4
#define CBSCALE 4096.0f    // rescale codebook into good f16 range

#define CHUNK        256                       // codes per LDS chunk
#define NCHUNK       (KCODE / CHUNK)           // 16 chunks per step
#define CHUNK_BYTES  (CHUNK * DLAT * 2)        // 32 KB
#define TOTCHUNK     (HSTEPS * NCHUNK)         // 64 chunks total

// ---------------------------------------------------------------------------
// Prep: f16 scaled codebook (entries ~uniform(-1,1)) + fp32 scaled ||e||^2.
// ---------------------------------------------------------------------------
__global__ __launch_bounds__(256) void vq_prep(const float* __restrict__ cb,
                                               _Float16* __restrict__ cbh,
                                               float* __restrict__ sesq) {
  int t = blockIdx.x * 256 + threadIdx.x;
  int w = t >> 5, lane = t & 31;            // w in [0, HSTEPS*KCODE)
  const float* row = cb + (size_t)w * DLAT;
  float e0 = row[2 * lane], e1 = row[2 * lane + 1];
  float ss = e0 * e0 + e1 * e1;
#pragma unroll
  for (int mask = 16; mask >= 1; mask >>= 1) ss += __shfl_xor(ss, mask, 32);
  _Float16* orow = cbh + (size_t)w * DLAT;
  orow[2 * lane]     = (_Float16)(e0 * CBSCALE);
  orow[2 * lane + 1] = (_Float16)(e1 * CBSCALE);
  if (lane == 0) sesq[w] = ss * CBSCALE;    // = CBSCALE * ||e||^2
}

// ---------------------------------------------------------------------------
// proj_in: r = z @ Wq + bq  (fp32 WMMA 16x16x4). One wave per 16-token tile,
// 4 accumulators cover all 64 output cols -> A fragment loaded once.
// ---------------------------------------------------------------------------
__global__ __launch_bounds__(256) void proj_in(const float* __restrict__ z,
                                               const float* __restrict__ Wq,
                                               const float* __restrict__ bq,
                                               float* __restrict__ r) {
  int t = blockIdx.x * 256 + threadIdx.x;
  int wave = t >> 5, lane = t & 31;
  int m = lane & 15, hi = lane >> 4;
  const float* zrow = z + (size_t)(wave * 16 + m) * INDIM;
  v8f acc[4] = {};
  for (int k0 = 0; k0 < INDIM; k0 += 4) {
    v2f a; a.x = zrow[k0 + 2 * hi]; a.y = zrow[k0 + 2 * hi + 1];
#pragma unroll
    for (int tn = 0; tn < 4; tn++) {
      v2f b;
      b.x = Wq[(size_t)(k0 + 2 * hi)     * DLAT + tn * 16 + m];
      b.y = Wq[(size_t)(k0 + 2 * hi + 1) * DLAT + tn * 16 + m];
      acc[tn] = __builtin_amdgcn_wmma_f32_16x16x4_f32(false, a, false, b, (short)0, acc[tn], false, false);
    }
  }
#pragma unroll
  for (int tn = 0; tn < 4; tn++) {
    float bias = bq[tn * 16 + m];
#pragma unroll
    for (int v = 0; v < 8; v++) {
      int tok = wave * 16 + v + 8 * hi;
      r[(size_t)tok * DLAT + tn * 16 + m] = acc[tn][v] + bias;
    }
  }
}

// ---------------------------------------------------------------------------
// vq_main: 4 residual-VQ steps; one wave per 16 tokens, residual + z_q held
// in registers across all steps. The f16 codebook is staged through a
// double-buffered 2x32KB LDS region with global_load_async_to_lds_b128
// (ASYNCcnt-tracked), shared by the 8 waves of the workgroup, prefetching
// chunk c+1 while WMMAs scan chunk c.
// ---------------------------------------------------------------------------
__global__ __launch_bounds__(256) void vq_main(const float* __restrict__ r,
                                               const float* __restrict__ cb,
                                               const _Float16* __restrict__ cbh,
                                               const float* __restrict__ sesq,
                                               float* __restrict__ zq) {
  __shared__ __attribute__((aligned(64))) _Float16 buf[2][CHUNK * DLAT]; // 64 KB
  int tid = threadIdx.x;
  int wl = tid >> 5, lane = tid & 31;
  int m = lane & 15, hi = lane >> 4;
  int tokBase = blockIdx.x * 128 + wl * 16;
  int segb[4] = { hi * 16, hi * 16 + 8, 32 + hi * 16, 40 + hi * 16 };

  // per-lane residual/accumulator registers in A-fragment order
  float rloc[32], zql[32];
  const float* rrow = r + (size_t)(tokBase + m) * DLAT;
#pragma unroll
  for (int s = 0; s < 4; s++)
#pragma unroll
    for (int j = 0; j < 8; j++) {
      rloc[s * 8 + j] = rrow[segb[s] + j];
      zql[s * 8 + j] = 0.0f;
    }

  unsigned lbase0 = (unsigned)(uintptr_t)(&buf[0][0]);
  unsigned lbase1 = (unsigned)(uintptr_t)(&buf[1][0]);

  // async-stage one 32KB chunk: 256 threads x 8 x b128 (16B) each
#define ISSUE_CHUNK(cg_, pb_)                                                  \
  do {                                                                         \
    const char* src_ = (const char*)cbh + (size_t)(cg_) * CHUNK_BYTES;         \
    unsigned lb_ = (pb_) ? lbase1 : lbase0;                                    \
    _Pragma("unroll")                                                          \
    for (int i_ = 0; i_ < CHUNK_BYTES / (256 * 16); ++i_) {                    \
      unsigned off_ = (unsigned)(i_ * 256 + tid) * 16u;                        \
      unsigned long long g_ = (unsigned long long)(uintptr_t)(src_ + off_);    \
      unsigned l_ = lb_ + off_;                                                \
      asm volatile("global_load_async_to_lds_b128 %0, %1, off"                 \
                   :: "v"(l_), "v"(g_) : "memory");                            \
    }                                                                          \
  } while (0)

  ISSUE_CHUNK(0, 0);
  asm volatile("s_wait_asynccnt 0x0" ::: "memory");
  __syncthreads();

  for (int h = 0; h < HSTEPS; h++) {
    // rebuild f16 A fragments from current residual registers
    v16h a0, a1;
#pragma unroll
    for (int i = 0; i < 16; i++) {
      a0[i] = (_Float16)rloc[i];
      a1[i] = (_Float16)rloc[16 + i];
    }
    float best[8]; int bidx[8];
#pragma unroll
    for (int v = 0; v < 8; v++) { best[v] = 3.0e38f; bidx[v] = 0; }
    const float* sesqH = sesq + h * KCODE;

    for (int cc = 0; cc < NCHUNK; ++cc) {
      int cg = h * NCHUNK + cc;
      int pb = cg & 1;
      if (cg + 1 < TOTCHUNK) ISSUE_CHUNK(cg + 1, (cg + 1) & 1);

      // scan 256 codes from LDS: 16 iters x (2 chained f16 WMMAs)
#pragma unroll 4
      for (int it = 0; it < CHUNK / 16; ++it) {
        int codeL = it * 16 + m;                 // code within chunk (B col)
        int code = cc * CHUNK + codeL;           // code within step
        const _Float16* brow = &buf[pb][codeL * DLAT + hi * 16];
        v16h b0 = *(const v16h*)(brow);          // K = hi*16..+15  (of 0..31)
        v16h b1 = *(const v16h*)(brow + 32);     // K = 32+hi*16..+15
        v8f c = {};
        c = __builtin_amdgcn_wmma_f32_16x16x32_f16(false, a0, false, b0, (short)0, c, false, false);
        c = __builtin_amdgcn_wmma_f32_16x16x32_f16(false, a1, false, b1, (short)0, c, false, false);
        float sq = sesqH[code];
#pragma unroll
        for (int v = 0; v < 8; v++) {
          float sc = sq - 2.0f * c[v];           // score for (row v+8hi, code)
          bool better = (sc < best[v]) || (sc == best[v] && code < bidx[v]);
          best[v] = better ? sc : best[v];
          bidx[v] = better ? code : bidx[v];
        }
      }
      asm volatile("s_wait_asynccnt 0x0" ::: "memory");
      __syncthreads();
    }

    // argmin across the 16 lanes of each half (rows 0-7 low, 8-15 high)
#pragma unroll
    for (int mask = 1; mask < 16; mask <<= 1) {
#pragma unroll
      for (int v = 0; v < 8; v++) {
        float ob = __shfl_xor(best[v], mask, 32);
        int   oi = __shfl_xor(bidx[v], mask, 32);
        bool better = (ob < best[v]) || (ob == best[v] && oi < bidx[v]);
        best[v] = better ? ob : best[v];
        bidx[v] = better ? oi : bidx[v];
      }
    }
    // route winning index to the lane that owns token row m = lane&15
    int srcLane = (m >> 3) << 4;
    int idxTok = 0;
#pragma unroll
    for (int v = 0; v < 8; v++) {
      int tv = __shfl(bidx[v], srcLane, 32);
      idxTok = ((m & 7) == v) ? tv : idxTok;
    }
    // gather fp32 codebook row; update residual + accumulator in registers
    const float* erow = cb + ((size_t)h * KCODE + idxTok) * DLAT;
#pragma unroll
    for (int s = 0; s < 4; s++)
#pragma unroll
      for (int j = 0; j < 8; j++) {
        float e = erow[segb[s] + j];
        rloc[s * 8 + j] -= e;
        zql[s * 8 + j] += e;
      }
  }
  float* zrow = zq + (size_t)(tokBase + m) * DLAT;
#pragma unroll
  for (int s = 0; s < 4; s++)
#pragma unroll
    for (int j = 0; j < 8; j++) zrow[segb[s] + j] = zql[s * 8 + j];
#undef ISSUE_CHUNK
}

// ---------------------------------------------------------------------------
// proj_out: out = zq @ Wp + bp  (fp32 WMMA 16x16x4). One wave per 16-token
// tile x 64-col group (4 accumulators share the A fragment).
// ---------------------------------------------------------------------------
__global__ __launch_bounds__(256) void proj_out(const float* __restrict__ zq,
                                                const float* __restrict__ Wp,
                                                const float* __restrict__ bp,
                                                float* __restrict__ out) {
  int t = blockIdx.x * 256 + threadIdx.x;
  int wave = t >> 5, lane = t & 31;
  int tile_m = wave >> 3;                   // 2048 token tiles
  int tg     = wave & 7;                    // 8 groups of 64 cols
  int m = lane & 15, hi = lane >> 4;
  const float* arow = zq + (size_t)(tile_m * 16 + m) * DLAT;
  v8f acc[4] = {};
#pragma unroll
  for (int k0 = 0; k0 < DLAT; k0 += 4) {
    v2f a; a.x = arow[k0 + 2 * hi]; a.y = arow[k0 + 2 * hi + 1];
#pragma unroll
    for (int tn = 0; tn < 4; tn++) {
      int col = tg * 64 + tn * 16 + m;
      v2f b;
      b.x = Wp[(size_t)(k0 + 2 * hi)     * INDIM + col];
      b.y = Wp[(size_t)(k0 + 2 * hi + 1) * INDIM + col];
      acc[tn] = __builtin_amdgcn_wmma_f32_16x16x4_f32(false, a, false, b, (short)0, acc[tn], false, false);
    }
  }
#pragma unroll
  for (int tn = 0; tn < 4; tn++) {
    int col = tg * 64 + tn * 16 + m;
    float bias = bp[col];
#pragma unroll
    for (int v = 0; v < 8; v++) {
      int tok = tile_m * 16 + v + 8 * hi;
      out[(size_t)tok * INDIM + col] = acc[tn][v] + bias;
    }
  }
}

// ---------------------------------------------------------------------------
extern "C" void kernel_launch(void* const* d_in, const int* in_sizes, int n_in,
                              void* d_out, int out_size, void* d_ws, size_t ws_size,
                              hipStream_t stream) {
  (void)in_sizes; (void)n_in; (void)out_size; (void)ws_size;
  const float* z  = (const float*)d_in[0];
  const float* Wq = (const float*)d_in[1];
  const float* bq = (const float*)d_in[2];
  const float* cb = (const float*)d_in[3];   // [4,4096,64] fp32
  const float* Wp = (const float*)d_in[4];
  const float* bp = (const float*)d_in[5];
  // d_in[6] = use_codebook_num (=4, static in reference; hardcoded as HSTEPS)
  float* out = (float*)d_out;

  char* ws = (char*)d_ws;
  size_t rBytes   = (size_t)NTOK * DLAT * sizeof(float);               // 8 MB
  size_t cbhBytes = (size_t)HSTEPS * KCODE * DLAT * sizeof(_Float16);  // 2 MB
  float*    r    = (float*)ws;
  float*    zqB  = (float*)(ws + rBytes);
  _Float16* cbh  = (_Float16*)(ws + 2 * rBytes);
  float*    sesq = (float*)(ws + 2 * rBytes + cbhBytes);

  vq_prep <<< (HSTEPS * KCODE * 32) / 256, 256, 0, stream >>> (cb, cbh, sesq);
  proj_in <<< (NTOK / 16) * 32 / 256,      256, 0, stream >>> (z, Wq, bq, r);
  vq_main <<< NTOK / 128,                  256, 0, stream >>> (r, cb, cbh, sesq, zqB);
  proj_out<<< (NTOK / 16) * 8 * 32 / 256,  256, 0, stream >>> (zqB, Wp, bp, out);
}
